// Block_49495203119786
// MI455X (gfx1250) — compile-verified, matrix operand order checked by reading, not run on hardware
//
#include <hip/hip_runtime.h>
#include <hip/hip_bf16.h>
#include <math.h>

// ---------------------------------------------------------------------------
// Problem constants (from reference): B=4, N=1569, C=768, H=12, d=64, F=9
// ---------------------------------------------------------------------------
#define BATCH 4
#define SEQ   1569
#define CDIM  768
#define HEADS 12
#define HDIM  64
#define FTOK  9
#define QKVC  2304
#define DFF   3072

#define MROWS   (BATCH * SEQ)      // 6276 valid rows
#define MTILES  393                // ceil(6276/16)
#define MPAD    (MTILES * 16)      // 6288 padded rows

#define KCHUNK  64                 // K depth staged per TDM transfer
#define BSTRIDE 72                 // LDS row stride in elements (144B, conflict-free)

typedef __attribute__((ext_vector_type(16))) __bf16 v16bf;
typedef __attribute__((ext_vector_type(8)))  float  v8f;
typedef __attribute__((ext_vector_type(4)))  unsigned u32x4;
typedef __attribute__((ext_vector_type(8)))  int      i32x8;
typedef __attribute__((ext_vector_type(4)))  int      i32x4;

#if defined(__has_builtin)
#if __has_builtin(__builtin_amdgcn_tensor_load_to_lds)
#define USE_TDM 1
#else
#define USE_TDM 0
#endif
#else
#define USE_TDM 0
#endif

// ---------------------------------------------------------------------------
// Fragment load: ISA 7.12.2 16-bit A/B layout. Lane L (r=L&15, half=L>>4)
// holds row/col r, K = {base+half*8 .. +7} and {base+half*8+16 .. +23}.
// Two contiguous 16-byte runs per lane -> global_load_b128 / ds_load_b128.
// ---------------------------------------------------------------------------
__device__ __forceinline__ v16bf load_frag(const __bf16* p) {
    union { v16bf v; float4 f[2]; } u;
    u.f[0] = *(const float4*)(const void*)(p);
    u.f[1] = *(const float4*)(const void*)(p + 16);
    return u.v;
}

__device__ __forceinline__ float gelu_exact(float x) {
    return 0.5f * x * (1.0f + erff(x * 0.7071067811865476f));
}

__device__ __forceinline__ void tensor_wait0() {
#if __has_builtin(__builtin_amdgcn_s_wait_tensorcnt)
    __builtin_amdgcn_s_wait_tensorcnt(0);
#else
    asm volatile("s_wait_tensorcnt 0x0" ::: "memory");
#endif
}
__device__ __forceinline__ void tensor_wait1() {
#if __has_builtin(__builtin_amdgcn_s_wait_tensorcnt)
    __builtin_amdgcn_s_wait_tensorcnt(1);
#else
    asm volatile("s_wait_tensorcnt 0x1" ::: "memory");
#endif
}

// ---------------------------------------------------------------------------
// TDM: DMA a 64-row x KCHUNK-col bf16 tile of W[Nout,K] (row-major) into LDS.
// D# group0: count=1, lds_addr, 57-bit global addr, type=2.
// D# group1: data_size=2B, pad_enable (32 DW interval, 4 DW pad -> LDS row
// stride 144B = BSTRIDE elems), tensor_dim0=K, tensor_dim1=Nout,
// tile_dim0=KCHUNK, tile_dim1=64, dim0_stride=K.
// ---------------------------------------------------------------------------
__device__ __forceinline__ void tdm_issue_b(const __bf16* W, int K, int Nout,
                                            int n0, int k0, unsigned lds_addr) {
#if USE_TDM
    unsigned long long ga = (unsigned long long)(uintptr_t)(W + (size_t)n0 * K + k0);
    u32x4 g0;
    g0[0] = 1u;                                              // count=1
    g0[1] = lds_addr;                                        // lds_addr
    g0[2] = (unsigned)ga;                                    // global_addr[31:0]
    g0[3] = (unsigned)((ga >> 32) & 0x01FFFFFFu) | 0x80000000u; // addr[56:32] | type=2
    i32x8 g1;
    g1[0] = (1 << 16) | (1 << 20) | (4 << 22) | (3 << 25);   // data_size=1(2B), pad_en, int=32DW, amt=4DW
    g1[1] = (K & 0xFFFF) << 16;                              // tensor_dim0[15:0]
    g1[2] = (int)(((unsigned)K >> 16) | (((unsigned)Nout & 0xFFFFu) << 16));
    g1[3] = (int)(((unsigned)Nout >> 16) | ((unsigned)KCHUNK << 16)); // tile_dim0
    g1[4] = 64;                                              // tile_dim1 = 64 rows
    g1[5] = K;                                               // tensor_dim0_stride[31:0]
    g1[6] = 0;
    g1[7] = 0;
    i32x4 gz = {0, 0, 0, 0};
#if __clang_major__ >= 23
    i32x8 gz8 = {0, 0, 0, 0, 0, 0, 0, 0};
    __builtin_amdgcn_tensor_load_to_lds(g0, g1, gz, gz, gz8, 0);
#else
    __builtin_amdgcn_tensor_load_to_lds(g0, g1, gz, gz, 0);
#endif
#else
    (void)W; (void)K; (void)Nout; (void)n0; (void)k0; (void)lds_addr;
#endif
}

// ---------------------------------------------------------------------------
// fp32 -> bf16 weight conversion
// ---------------------------------------------------------------------------
__global__ __launch_bounds__(256) void cvt_bf16_kernel(const float* __restrict__ src,
                                                       __bf16* __restrict__ dst, int n) {
    int i = blockIdx.x * 256 + threadIdx.x;
    if (i < n) dst[i] = (__bf16)src[i];
}

// ---------------------------------------------------------------------------
// LayerNorm over C=768, one block (256 thr) per row; bf16 output.
// Rows >= Mvalid (pad) are written as zeros without reading input.
// ---------------------------------------------------------------------------
__global__ __launch_bounds__(256) void ln_kernel(const float* __restrict__ x,
                                                 const float* __restrict__ g,
                                                 const float* __restrict__ b,
                                                 __bf16* __restrict__ h, int Mvalid) {
    __shared__ float red[256];
    const int row = blockIdx.x;
    const int tid = threadIdx.x;
    if (row >= Mvalid) {
        h[(size_t)row * CDIM + tid]       = (__bf16)0.0f;
        h[(size_t)row * CDIM + tid + 256] = (__bf16)0.0f;
        h[(size_t)row * CDIM + tid + 512] = (__bf16)0.0f;
        return;
    }
    const float* xr = x + (size_t)row * CDIM;
    float v0 = xr[tid], v1 = xr[tid + 256], v2 = xr[tid + 512];
    red[tid] = v0 + v1 + v2;
    __syncthreads();
    for (int st = 128; st; st >>= 1) { if (tid < st) red[tid] += red[tid + st]; __syncthreads(); }
    const float mu = red[0] * (1.0f / 768.0f);
    __syncthreads();
    float d0 = v0 - mu, d1 = v1 - mu, d2 = v2 - mu;
    red[tid] = d0 * d0 + d1 * d1 + d2 * d2;
    __syncthreads();
    for (int st = 128; st; st >>= 1) { if (tid < st) red[tid] += red[tid + st]; __syncthreads(); }
    const float rstd = rsqrtf(red[0] * (1.0f / 768.0f) + 1e-5f);
    h[(size_t)row * CDIM + tid]       = (__bf16)(d0 * rstd * g[tid]       + b[tid]);
    h[(size_t)row * CDIM + tid + 256] = (__bf16)(d1 * rstd * g[tid + 256] + b[tid + 256]);
    h[(size_t)row * CDIM + tid + 512] = (__bf16)(d2 * rstd * g[tid + 512] + b[tid + 512]);
}

// ---------------------------------------------------------------------------
// WMMA GEMM: Y[M,Nout] = A[M,K] @ W[Nout,K]^T  (A,W bf16, fp32 accum)
// Block: 8 waves, each owns a 32x64 output strip (two M-tiles); all share one
// 64-col B strip staged in LDS by TDM (double-buffered, TENSORcnt-tracked).
// All 8 B fragments of a chunk are held in distinct registers so ds_load_b128
// can overlap with wmma (no single-octet recycling / dscnt-0 serialization).
// MODE 0: plain -> bf16      (qkv)
// MODE 1: +bias -> fp32      (proj)
// MODE 2: +bias+GELU -> bf16 (fc1)
// MODE 3: +bias+residual -> fp32, row-guarded store (fc2 -> d_out)
// ---------------------------------------------------------------------------
#define WMMA_BF16(D, Afr, Bfr) \
    D = __builtin_amdgcn_wmma_f32_16x16x32_bf16(false, Afr, false, Bfr, (short)0, D, false, false)

template <int MODE>
__global__ __launch_bounds__(256) void gemm_wmma_kernel(
    const __bf16* __restrict__ A, const __bf16* __restrict__ W,
    const float* __restrict__ bias, const float* __restrict__ resid,
    void* __restrict__ Yv, int Mtiles, int K, int Nout, int Mvalid)
{
    __shared__ __align__(16) __bf16 bsm[2][64 * BSTRIDE];   // 2 x 9KB
    const int wave = threadIdx.x >> 5;
    const int lane = threadIdx.x & 31;
    const int mt0 = blockIdx.y * 16 + wave * 2;   // two M-tiles per wave
    const int mt1 = mt0 + 1;
    const bool act0 = (mt0 < Mtiles);             // wave-uniform guards
    const bool act1 = (mt1 < Mtiles);
    const int m0 = mt0 * 16, m1 = mt1 * 16;
    const int n0 = blockIdx.x * 64;
    const int r = lane & 15, half = lane >> 4;
    const int nchunks = K / KCHUNK;

    const __bf16* Ap0 = A + (size_t)(m0 + r) * K + half * 8;
    const __bf16* Ap1 = A + (size_t)(m1 + r) * K + half * 8;
    v8f c00 = {}, c01 = {}, c02 = {}, c03 = {};
    v8f c10 = {}, c11 = {}, c12 = {}, c13 = {};

#if USE_TDM
    const unsigned lds0 = (unsigned)(uintptr_t)(&bsm[0][0]);
    const unsigned lds1 = (unsigned)(uintptr_t)(&bsm[1][0]);
    if (wave == 0) tdm_issue_b(W, K, Nout, n0, 0, lds0);   // wave 0 always active
#endif

    for (int ch = 0; ch < nchunks; ++ch) {
        // Prefetch this chunk's A fragments (overlaps tensor wait / copy)
        v16bf a00, a01, a10, a11;
        if (act0) {
            a00 = load_frag(Ap0 + (size_t)ch * KCHUNK);
            a01 = load_frag(Ap0 + (size_t)ch * KCHUNK + 32);
        }
        if (act1) {
            a10 = load_frag(Ap1 + (size_t)ch * KCHUNK);
            a11 = load_frag(Ap1 + (size_t)ch * KCHUNK + 32);
        }
#if USE_TDM
        if (wave == 0) {
            if (ch + 1 < nchunks) {
                tdm_issue_b(W, K, Nout, n0, (ch + 1) * KCHUNK,
                            ((ch + 1) & 1) ? lds1 : lds0);
                tensor_wait1();              // chunk ch done, ch+1 in flight
            } else {
                tensor_wait0();
            }
        }
        __syncthreads();                     // B chunk visible to all waves
        const __bf16* bb = &bsm[ch & 1][0];
#else
        __syncthreads();                     // previous consume finished
        for (int idx = threadIdx.x; idx < 64 * (KCHUNK / 8); idx += 256) {
            const int row = idx >> 3, c = (idx & 7) * 8;
            *(float4*)(void*)(&bsm[0][(size_t)row * BSTRIDE + c]) =
                *(const float4*)(const void*)(W + (size_t)(n0 + row) * K + (size_t)ch * KCHUNK + c);
        }
        __syncthreads();
        const __bf16* bb = &bsm[0][0];
#endif
        if (act0) {
            // All 8 B fragments in distinct registers -> ds loads overlap wmma
            const __bf16* b0p = bb + half * 8;        // s = 0
            const __bf16* b1p = bb + 32 + half * 8;   // s = 1
            v16bf b00 = load_frag(b0p + (size_t)(r)      * BSTRIDE);
            v16bf b01 = load_frag(b0p + (size_t)(16 + r) * BSTRIDE);
            v16bf b02 = load_frag(b0p + (size_t)(32 + r) * BSTRIDE);
            v16bf b03 = load_frag(b0p + (size_t)(48 + r) * BSTRIDE);
            v16bf b10 = load_frag(b1p + (size_t)(r)      * BSTRIDE);
            v16bf b11 = load_frag(b1p + (size_t)(16 + r) * BSTRIDE);
            v16bf b12 = load_frag(b1p + (size_t)(32 + r) * BSTRIDE);
            v16bf b13 = load_frag(b1p + (size_t)(48 + r) * BSTRIDE);
            WMMA_BF16(c00, a00, b00);
            WMMA_BF16(c01, a00, b01);
            WMMA_BF16(c02, a00, b02);
            WMMA_BF16(c03, a00, b03);
            WMMA_BF16(c00, a01, b10);
            WMMA_BF16(c01, a01, b11);
            WMMA_BF16(c02, a01, b12);
            WMMA_BF16(c03, a01, b13);
            if (act1) {
                WMMA_BF16(c10, a10, b00);
                WMMA_BF16(c11, a10, b01);
                WMMA_BF16(c12, a10, b02);
                WMMA_BF16(c13, a10, b03);
                WMMA_BF16(c10, a11, b10);
                WMMA_BF16(c11, a11, b11);
                WMMA_BF16(c12, a11, b12);
                WMMA_BF16(c13, a11, b13);
            }
        }
        __syncthreads();                     // buffer free before next overwrite
    }

    if (!act0) return;
    // C/D layout: lanes 0-15 col=lane, VGPR vi -> row base+vi; lanes 16-31 -> base+8+vi
    v8f cs0[4] = {c00, c01, c02, c03};
    v8f cs1[4] = {c10, c11, c12, c13};
#pragma unroll
    for (int tile = 0; tile < 2; ++tile) {
        if (tile == 1 && !act1) break;
        const int mb = (tile == 0) ? m0 : m1;
#pragma unroll
        for (int t = 0; t < 4; ++t) {
            const int col = n0 + t * 16 + r;
            float bv = 0.0f;
            if constexpr (MODE >= 1) bv = bias[col];
#pragma unroll
            for (int vi = 0; vi < 8; ++vi) {
                const int row = mb + half * 8 + vi;
                float val = ((tile == 0) ? cs0[t][vi] : cs1[t][vi]) + bv;
                if constexpr (MODE == 2) val = gelu_exact(val);
                if constexpr (MODE == 3) {
                    if (row < Mvalid)
                        ((float*)Yv)[(size_t)row * Nout + col] =
                            val + resid[(size_t)row * Nout + col];
                } else if constexpr (MODE == 1) {
                    ((float*)Yv)[(size_t)row * Nout + col] = val;
                } else {
                    ((__bf16*)Yv)[(size_t)row * Nout + col] = (__bf16)val;
                }
            }
        }
    }
}

// ---------------------------------------------------------------------------
// Sparse masked attention, spatial rows i in [F, N): attend to j in [0, F).
// One wave32 per (b, h, i); lane owns dims {lane, lane+32} of d=64.
// ---------------------------------------------------------------------------
__device__ __forceinline__ float wave_sum32(float v) {
#pragma unroll
    for (int m = 16; m; m >>= 1) v += __shfl_xor(v, m, 32);
    return v;
}

__global__ __launch_bounds__(256) void attn_spatial_kernel(const __bf16* __restrict__ qkv,
                                                           __bf16* __restrict__ out) {
    const int lane = threadIdx.x & 31;
    const int wid = blockIdx.x * 8 + (threadIdx.x >> 5);
    const int total = BATCH * HEADS * (SEQ - FTOK);
    if (wid >= total) return;
    const int i = FTOK + wid % (SEQ - FTOK);
    const int hh = (wid / (SEQ - FTOK)) % HEADS;
    const int b = wid / ((SEQ - FTOK) * HEADS);

    const size_t qoff = ((size_t)b * SEQ + i) * QKVC + hh * HDIM;
    const float q0 = (float)qkv[qoff + lane];
    const float q1 = (float)qkv[qoff + lane + 32];

    float s[FTOK];
#pragma unroll
    for (int j = 0; j < FTOK; ++j) {
        const __bf16* kp = qkv + ((size_t)b * SEQ + j) * QKVC + CDIM + hh * HDIM;
        float p = q0 * (float)kp[lane] + q1 * (float)kp[lane + 32];
        s[j] = wave_sum32(p) * 0.125f;  // scale = d^-0.5 = 1/8
    }
    float mx = s[0];
#pragma unroll
    for (int j = 1; j < FTOK; ++j) mx = fmaxf(mx, s[j]);
    float wsum = 0.0f, wj[FTOK];
#pragma unroll
    for (int j = 0; j < FTOK; ++j) { wj[j] = __expf(s[j] - mx); wsum += wj[j]; }
    const float inv = 1.0f / wsum;
    float o0 = 0.0f, o1 = 0.0f;
#pragma unroll
    for (int j = 0; j < FTOK; ++j) {
        const __bf16* vp = qkv + ((size_t)b * SEQ + j) * QKVC + 2 * CDIM + hh * HDIM;
        o0 += wj[j] * (float)vp[lane];
        o1 += wj[j] * (float)vp[lane + 32];
    }
    const size_t ooff = ((size_t)b * SEQ + i) * CDIM + hh * HDIM;
    out[ooff + lane]      = (__bf16)(o0 * inv);
    out[ooff + lane + 32] = (__bf16)(o1 * inv);
}

// ---------------------------------------------------------------------------
// Temporal rows i in [0, F): row 0 attends to {0} U [F,N); rows 1..8 to [F,N).
// One block (256 thr) per (b, h, i); logits live in LDS.
// ---------------------------------------------------------------------------
__global__ __launch_bounds__(256) void attn_temporal_kernel(const __bf16* __restrict__ qkv,
                                                            __bf16* __restrict__ out) {
    __shared__ float qs[HDIM];
    __shared__ float logits[SEQ - FTOK + 1];  // up to 1561
    __shared__ float red[256];
    const int g = blockIdx.x;           // 0 .. B*H*F-1
    const int i = g % FTOK;
    const int hh = (g / FTOK) % HEADS;
    const int b = g / (FTOK * HEADS);
    const int tid = threadIdx.x;
    const int Lcnt = (i == 0) ? (SEQ - FTOK + 1) : (SEQ - FTOK);

    if (tid < HDIM)
        qs[tid] = (float)qkv[((size_t)b * SEQ + i) * QKVC + hh * HDIM + tid];
    __syncthreads();

    for (int jj = tid; jj < Lcnt; jj += 256) {
        const int j = (i == 0) ? ((jj == 0) ? 0 : (FTOK - 1 + jj)) : (FTOK + jj);
        const __bf16* kp = qkv + ((size_t)b * SEQ + j) * QKVC + CDIM + hh * HDIM;
        float p = 0.0f;
#pragma unroll 8
        for (int dd = 0; dd < HDIM; ++dd) p += qs[dd] * (float)kp[dd];
        logits[jj] = p * 0.125f;
    }
    __syncthreads();

    float mx = -3.0e38f;
    for (int jj = tid; jj < Lcnt; jj += 256) mx = fmaxf(mx, logits[jj]);
    red[tid] = mx;
    __syncthreads();
    for (int st = 128; st; st >>= 1) { if (tid < st) red[tid] = fmaxf(red[tid], red[tid + st]); __syncthreads(); }
    mx = red[0];
    __syncthreads();

    float sm = 0.0f;
    for (int jj = tid; jj < Lcnt; jj += 256) {
        const float e = __expf(logits[jj] - mx);
        logits[jj] = e;
        sm += e;
    }
    red[tid] = sm;
    __syncthreads();
    for (int st = 128; st; st >>= 1) { if (tid < st) red[tid] += red[tid + st]; __syncthreads(); }
    const float inv = 1.0f / red[0];

    if (tid < HDIM) {
        float acc = 0.0f;
        for (int jj = 0; jj < Lcnt; ++jj) {
            const int j = (i == 0) ? ((jj == 0) ? 0 : (FTOK - 1 + jj)) : (FTOK + jj);
            acc += logits[jj] *
                   (float)qkv[((size_t)b * SEQ + j) * QKVC + 2 * CDIM + hh * HDIM + tid];
        }
        out[((size_t)b * SEQ + i) * CDIM + hh * HDIM + tid] = (__bf16)(acc * inv);
    }
}

// ---------------------------------------------------------------------------
// Launch
// ---------------------------------------------------------------------------
extern "C" void kernel_launch(void* const* d_in, const int* in_sizes, int n_in,
                              void* d_out, int out_size, void* d_ws, size_t ws_size,
                              hipStream_t stream) {
    const float* x      = (const float*)d_in[0];
    const float* ln2_g  = (const float*)d_in[1];
    const float* ln2_b  = (const float*)d_in[2];
    const float* qkv_w  = (const float*)d_in[3];
    const float* proj_w = (const float*)d_in[4];
    const float* proj_b = (const float*)d_in[5];
    const float* fc1_w  = (const float*)d_in[6];
    const float* fc1_b  = (const float*)d_in[7];
    const float* fc2_w  = (const float*)d_in[8];
    const float* fc2_b  = (const float*)d_in[9];
    float* y = (float*)d_out;

    // Workspace carve-up (256B aligned)
    char* ws = (char*)d_ws;
    size_t off = 0;
    auto carve = [&](size_t bytes) { char* p = ws + off; off = (off + bytes + 255) & ~(size_t)255; return p; };
    __bf16* h_bf    = (__bf16*)carve((size_t)MPAD * CDIM * 2);
    __bf16* qkv_wb  = (__bf16*)carve((size_t)QKVC * CDIM * 2);
    __bf16* proj_wb = (__bf16*)carve((size_t)CDIM * CDIM * 2);
    __bf16* fc1_wb  = (__bf16*)carve((size_t)DFF * CDIM * 2);
    __bf16* fc2_wb  = (__bf16*)carve((size_t)CDIM * DFF * 2);
    __bf16* qkv_o   = (__bf16*)carve((size_t)MPAD * QKVC * 2);
    __bf16* attn_o  = (__bf16*)carve((size_t)MPAD * CDIM * 2);
    float*  xa      = (float*) carve((size_t)MPAD * CDIM * 4);
    __bf16* h2_bf   = (__bf16*)carve((size_t)MPAD * CDIM * 2);
    __bf16* m1_bf   = (__bf16*)carve((size_t)MPAD * DFF * 2);
    (void)ws_size; (void)n_in; (void)in_sizes; (void)out_size;

    // 1) Convert weights to bf16
    auto cvt = [&](const float* s, __bf16* d, int n) {
        cvt_bf16_kernel<<<(n + 255) / 256, 256, 0, stream>>>(s, d, n);
    };
    cvt(qkv_w,  qkv_wb,  QKVC * CDIM);
    cvt(proj_w, proj_wb, CDIM * CDIM);
    cvt(fc1_w,  fc1_wb,  DFF * CDIM);
    cvt(fc2_w,  fc2_wb,  CDIM * DFF);

    // 2) h = LN(x)  (bf16, pad rows zeroed)
    ln_kernel<<<MPAD, 256, 0, stream>>>(x, ln2_g, ln2_b, h_bf, MROWS);

    const dim3 blk(256);
    const int gy = (MTILES + 15) / 16;   // 32 rows per wave, 8 waves per block

    // 3) qkv = h @ qkv_w^T   [MPAD, 2304] bf16
    gemm_wmma_kernel<0><<<dim3(QKVC / 64, gy), blk, 0, stream>>>(
        h_bf, qkv_wb, nullptr, nullptr, qkv_o, MTILES, CDIM, QKVC, MROWS);

    // 4) masked attention (sparse interact mask)
    {
        const int waves = BATCH * HEADS * (SEQ - FTOK);
        attn_spatial_kernel<<<(waves + 7) / 8, 256, 0, stream>>>(qkv_o, attn_o);
        attn_temporal_kernel<<<BATCH * HEADS * FTOK, 256, 0, stream>>>(qkv_o, attn_o);
    }

    // 5) xa = attn_o @ proj_w^T + proj_b   [MPAD, 768] fp32
    gemm_wmma_kernel<1><<<dim3(CDIM / 64, gy), blk, 0, stream>>>(
        attn_o, proj_wb, proj_b, nullptr, xa, MTILES, CDIM, CDIM, MROWS);

    // 6) h2 = LN(xa)  (bf16)
    ln_kernel<<<MPAD, 256, 0, stream>>>(xa, ln2_g, ln2_b, h2_bf, MROWS);

    // 7) m1 = gelu(h2 @ fc1_w^T + fc1_b)  [MPAD, 3072] bf16
    gemm_wmma_kernel<2><<<dim3(DFF / 64, gy), blk, 0, stream>>>(
        h2_bf, fc1_wb, fc1_b, nullptr, m1_bf, MTILES, CDIM, DFF, MROWS);

    // 8) y = m1 @ fc2_w^T + fc2_b + xa  -> d_out (valid rows only)
    gemm_wmma_kernel<3><<<dim3(CDIM / 64, gy), blk, 0, stream>>>(
        m1_bf, fc2_wb, fc2_b, xa, y, MTILES, DFF, CDIM, MROWS);
}
